// SwinTransformerBlock2D_72713796321802
// MI455X (gfx1250) — matrix operand order repaired and checked by reading, not compiled, not run on hardware
//
#include <hip/hip_runtime.h>
#include <hip/hip_bf16.h>

typedef __attribute__((ext_vector_type(16))) _Float16 v16h;
typedef __attribute__((ext_vector_type(8)))  _Float16 v8h;
typedef __attribute__((ext_vector_type(8)))  float    v8f;

#define DIMC 192
#define HEADS 6
#define WSZ 7
#define SHIFT_SZ 3
#define HD 32
#define NTOK 49
#define HW 56
#define QK_SCALE 0.17677669529663687f

__device__ __forceinline__ v8f wmma16(v16h a, v16h b, v8f c) {
  // D = A(16x32 f16) * B(32x16 f16) + C(16x16 f32)
  return __builtin_amdgcn_wmma_f32_16x16x32_f16(false, a, false, b, (short)0, c, false, false);
}

// A fragment: rows row0..row0+15, K k0..k0+31, from row-major [row][k] buffer.
// lane 0-15: halves0-15=K(k0..k0+15)
// lane16-31: halves0-15=K(k0+16..k0+31)
__device__ __forceinline__ v16h ld_afrag(const _Float16* base, int stride, int row0, int k0, int lane) {
  const int r = lane & 15, hi = lane >> 4;
  const _Float16* p = base + (row0 + r) * stride + k0 + hi * 16;
  v8h lo = *(const v8h*)p;
  v8h hj = *(const v8h*)(p + 8);
  v16h a;
#pragma unroll
  for (int i = 0; i < 8; ++i) { a[i] = lo[i]; a[i + 8] = hj[i]; }
  return a;
}

// B fragment from [N][K] (K-contiguous) buffer: lane n=n0+(l&15),
// halves 0..15 = K(k0 + (l>=16?16:0) .. +15) -> one contiguous 32B load.
__device__ __forceinline__ v16h ld_bfrag(const _Float16* W, int kdim, int n0, int k0, int lane) {
  const int c = lane & 15, hi = lane >> 4;
  return *(const v16h*)(W + (n0 + c) * kdim + k0 + hi * 16);
}

__global__ void __launch_bounds__(256) convert_weights_kernel(
    const float* __restrict__ qkv_w, const float* __restrict__ proj_w,
    const float* __restrict__ fc1_w, const float* __restrict__ fc2_w,
    _Float16* __restrict__ qkvT, _Float16* __restrict__ projT,
    _Float16* __restrict__ fc1T, _Float16* __restrict__ fc2T)
{
  const int i = blockIdx.x * 256 + threadIdx.x;
  if (i < 576 * 192) { const int n = i / 192, k = i % 192; qkvT[i] = (_Float16)qkv_w[k * 576 + n]; }
  if (i < 192 * 192) { const int n = i / 192, k = i % 192; projT[i] = (_Float16)proj_w[k * 192 + n]; }
  if (i < 768 * 192) { const int n = i / 192, k = i % 192; fc1T[i]  = (_Float16)fc1_w[k * 768 + n]; }
  if (i < 192 * 768) { const int n = i / 768, k = i % 768; fc2T[i]  = (_Float16)fc2_w[k * 192 + n]; }
}

__device__ __forceinline__ int region_id(int yr, int xr) {
  const int ry = yr < (HW - WSZ) ? 0 : (yr < (HW - SHIFT_SZ) ? 1 : 2);
  const int rx = xr < (HW - WSZ) ? 0 : (xr < (HW - SHIFT_SZ) ? 1 : 2);
  return ry * 3 + rx;
}

// One workgroup per window: 16 batches * 8 * 8 windows = 1024 blocks, 256 threads.
__global__ void __launch_bounds__(256) swin_attn_kernel(
    const float* __restrict__ xin, const float* __restrict__ n1g, const float* __restrict__ n1b,
    const _Float16* __restrict__ qkvT, const float* __restrict__ qkv_b,
    const _Float16* __restrict__ projT, const float* __restrict__ proj_b,
    const float* __restrict__ rel_bias, float* __restrict__ x1)
{
  __shared__ __align__(32) _Float16 sA[64 * DIMC];   // 24KB: LN'd window tokens (rows 49..63 = 0)
  __shared__ __align__(32) _Float16 sQ[64 * HD];     // 4KB  per-head q (scaled)
  __shared__ __align__(32) _Float16 sK[64 * HD];     // 4KB  per-head k
  __shared__ __align__(32) _Float16 sVT[HD * 64];    // 4KB  per-head v transposed [d][token]
  __shared__ __align__(32) float    sScore[64 * 64]; // 16KB scores f32, rewritten as f16 probs
  __shared__ __align__(32) _Float16 sAV[64 * HD];    // 4KB  attn@v per head

  const int tid = threadIdx.x;
  const int lane = tid & 31, wv = tid >> 5;
  const int widx = blockIdx.x;
  const int bb = widx >> 6;
  const int wy = (widx >> 3) & 7;
  const int wx = widx & 7;

  // ---- LN1 + shifted-window gather: 4 lanes per token ----
  {
    const int t = tid >> 2, sub = tid & 3;
    if (t < NTOK) {
      const int ty = t / WSZ, tx = t % WSZ;
      const int gy = (wy * WSZ + ty + SHIFT_SZ) % HW;
      const int gx = (wx * WSZ + tx + SHIFT_SZ) % HW;
      const int base = (bb * DIMC) * (HW * HW) + gy * HW + gx;
      float vals[48];
      float s = 0.f, s2 = 0.f;
#pragma unroll
      for (int j = 0; j < 48; ++j) {
        const int c = sub + 4 * j;
        const float v = xin[base + c * (HW * HW)];
        vals[j] = v; s += v; s2 += v * v;
      }
      s  += __shfl_xor(s, 1);  s  += __shfl_xor(s, 2);
      s2 += __shfl_xor(s2, 1); s2 += __shfl_xor(s2, 2);
      const float mu = s * (1.f / DIMC);
      const float rs = rsqrtf(s2 * (1.f / DIMC) - mu * mu + 1e-5f);
#pragma unroll
      for (int j = 0; j < 48; ++j) {
        const int c = sub + 4 * j;
        sA[t * DIMC + c] = (_Float16)((vals[j] - mu) * rs * n1g[c] + n1b[c]);
      }
    } else if (t < 64) {
#pragma unroll
      for (int j = 0; j < 48; ++j) sA[t * DIMC + sub + 4 * j] = (_Float16)0.f;
    }
  }
  __syncthreads();

  v8f pacc[6];  // projection accumulators: 48 output tiles / 8 waves
#pragma unroll
  for (int i = 0; i < 6; ++i) { v8f z = {}; pacc[i] = z; }

  for (int h = 0; h < HEADS; ++h) {
    // --- QKV for head h: q,k,v each 64x32; 24 tiles over 8 waves ---
#pragma unroll
    for (int i = 0; i < 3; ++i) {
      const int t = wv * 3 + i;        // 0..23
      const int mat = t >> 3;          // 0=q 1=k 2=v
      const int rem = t & 7;
      const int mt = rem >> 1, nt2 = rem & 1;
      const int col0 = mat * DIMC + h * HD + nt2 * 16;
      v8f acc = {};
      for (int kk = 0; kk < 6; ++kk) {
        v16h a = ld_afrag(sA, DIMC, mt * 16, kk * 32, lane);
        v16h b = ld_bfrag(qkvT, DIMC, col0, kk * 32, lane);
        acc = wmma16(a, b, acc);
      }
      const int cl = nt2 * 16 + (lane & 15);
      const float bias = qkv_b[col0 + (lane & 15)];
      const int hi = lane >> 4;
#pragma unroll
      for (int r = 0; r < 8; ++r) {
        const int row = mt * 16 + r + hi * 8;
        const float v = acc[r] + bias;
        if (mat == 0)      sQ[row * HD + cl] = (_Float16)(v * QK_SCALE);
        else if (mat == 1) sK[row * HD + cl] = (_Float16)v;
        else               sVT[cl * 64 + row] = (_Float16)v;
      }
    }
    __syncthreads();

    // --- scores = q @ k^T + rel_bias + shift mask ---
#pragma unroll
    for (int i = 0; i < 2; ++i) {
      const int t = wv * 2 + i;        // 0..15
      const int mt = t >> 2, nt = t & 3;
      v16h a = ld_afrag(sQ, HD, mt * 16, 0, lane);
      v16h b = ld_bfrag(sK, HD, nt * 16, 0, lane);
      v8f acc = {};
      acc = wmma16(a, b, acc);
      const int hi = lane >> 4;
      const int kj = nt * 16 + (lane & 15);
      int kid = 0, ky = 0, kx = 0;
      if (kj < NTOK) {
        ky = kj / WSZ; kx = kj % WSZ;
        kid = region_id(wy * WSZ + ky, wx * WSZ + kx);
      }
#pragma unroll
      for (int r = 0; r < 8; ++r) {
        const int qi = mt * 16 + r + hi * 8;
        float s = -1e30f;
        if (kj < NTOK && qi < NTOK) {
          const int qy = qi / WSZ, qx = qi % WSZ;
          if (region_id(wy * WSZ + qy, wx * WSZ + qx) == kid) {
            const int ridx = (qy - ky + 6) * 13 + (qx - kx + 6);
            s = acc[r] + rel_bias[ridx * HEADS + h];
          }
        }
        sScore[qi * 64 + kj] = s;
      }
    }
    __syncthreads();

    // --- softmax (one thread per row); rewrite row in place as f16 probs ---
    if (tid < 64) {
      float buf[64];
      float m = -1e30f;
#pragma unroll
      for (int j = 0; j < 64; ++j) { buf[j] = sScore[tid * 64 + j]; m = fmaxf(m, buf[j]); }
      float sum = 0.f;
#pragma unroll
      for (int j = 0; j < 64; ++j) { buf[j] = __expf(buf[j] - m); sum += buf[j]; }
      const float inv = 1.f / sum;
      _Float16* prow = (_Float16*)&sScore[tid * 64];
#pragma unroll
      for (int j = 0; j < 64; ++j) prow[j] = (_Float16)(buf[j] * inv);
    }
    __syncthreads();

    // --- attn @ v : 8 tiles, one per wave ---
    {
      const int mt = wv >> 1, nt = wv & 1;
      v8f acc = {};
      for (int kk = 0; kk < 2; ++kk) {
        v16h a = ld_afrag((const _Float16*)sScore, 128, mt * 16, kk * 32, lane);
        v16h b = ld_bfrag(sVT, 64, nt * 16, kk * 32, lane);
        acc = wmma16(a, b, acc);
      }
      const int hi = lane >> 4;
      const int d = nt * 16 + (lane & 15);
#pragma unroll
      for (int r = 0; r < 8; ++r) sAV[(mt * 16 + r + hi * 8) * HD + d] = (_Float16)acc[r];
    }
    __syncthreads();

    // --- projection partial: accumulate (attn@v)_h @ proj_w[h*32 : h*32+32, :] ---
#pragma unroll
    for (int i = 0; i < 6; ++i) {
      const int gt = wv * 6 + i;       // 0..47
      const int mt = gt / 12, nt = gt % 12;
      v16h a = ld_afrag(sAV, HD, mt * 16, 0, lane);
      v16h b = ld_bfrag(projT, DIMC, nt * 16, h * HD, lane);
      pacc[i] = wmma16(a, b, pacc[i]);
    }
    __syncthreads();
  }

  // --- un-shift scatter: x1 = shortcut + proj_out + proj_b ---
#pragma unroll
  for (int i = 0; i < 6; ++i) {
    const int gt = wv * 6 + i;
    const int mt = gt / 12, nt = gt % 12;
    const int c = nt * 16 + (lane & 15);
    const float pb = proj_b[c];
    const int hi = lane >> 4;
#pragma unroll
    for (int r = 0; r < 8; ++r) {
      const int row = mt * 16 + r + hi * 8;
      if (row < NTOK) {
        const int ty = row / WSZ, tx = row % WSZ;
        const int gy = (wy * WSZ + ty + SHIFT_SZ) % HW;
        const int gx = (wx * WSZ + tx + SHIFT_SZ) % HW;
        const int idx = ((bb * DIMC + c) * HW + gy) * HW + gx;
        x1[idx] = xin[idx] + pacc[i][r] + pb;
      }
    }
  }
}

// 784 workgroups x 64 tokens: LN2 + fc1/GELU + fc2 + residual
__global__ void __launch_bounds__(256) swin_mlp_kernel(
    const float* __restrict__ x1, const float* __restrict__ n2g, const float* __restrict__ n2b,
    const _Float16* __restrict__ fc1T, const float* __restrict__ fc1_b,
    const _Float16* __restrict__ fc2T, const float* __restrict__ fc2_b,
    float* __restrict__ out)
{
  __shared__ __align__(32) _Float16 sA[64 * DIMC]; // 24KB
  __shared__ __align__(32) _Float16 sH[64 * 96];   // 12KB staging for fc1 chunk

  const int tid = threadIdx.x;
  const int lane = tid & 31, wv = tid >> 5;
  const int tok0 = blockIdx.x * 64;

  {
    const int tl = tid >> 2, sub = tid & 3;
    const int tok = tok0 + tl;
    const int b = tok / 3136, yx = tok % 3136;
    const int base = b * (DIMC * 3136) + yx;
    float vals[48];
    float s = 0.f, s2 = 0.f;
#pragma unroll
    for (int j = 0; j < 48; ++j) {
      const int c = sub + 4 * j;
      const float v = x1[base + c * 3136];
      vals[j] = v; s += v; s2 += v * v;
    }
    s  += __shfl_xor(s, 1);  s  += __shfl_xor(s, 2);
    s2 += __shfl_xor(s2, 1); s2 += __shfl_xor(s2, 2);
    const float mu = s * (1.f / DIMC);
    const float rs = rsqrtf(s2 * (1.f / DIMC) - mu * mu + 1e-5f);
#pragma unroll
    for (int j = 0; j < 48; ++j) {
      const int c = sub + 4 * j;
      sA[tl * DIMC + c] = (_Float16)((vals[j] - mu) * rs * n2g[c] + n2b[c]);
    }
  }
  __syncthreads();

  v8f facc[6];
#pragma unroll
  for (int i = 0; i < 6; ++i) { v8f z = {}; facc[i] = z; }

  for (int cc = 0; cc < 8; ++cc) {     // 8 chunks of 96 hidden cols (768 total)
    const int col0 = cc * 96;
#pragma unroll
    for (int i = 0; i < 3; ++i) {
      const int t = wv * 3 + i;        // 24 fc1 tiles
      const int mt = t / 6, nt = t % 6;
      const int col = col0 + nt * 16;
      v8f acc = {};
      for (int kk = 0; kk < 6; ++kk) {
        v16h a = ld_afrag(sA, DIMC, mt * 16, kk * 32, lane);
        v16h b = ld_bfrag(fc1T, DIMC, col, kk * 32, lane);
        acc = wmma16(a, b, acc);
      }
      const float bias = fc1_b[col + (lane & 15)];
      const int hi = lane >> 4, cl = nt * 16 + (lane & 15);
#pragma unroll
      for (int r = 0; r < 8; ++r) {
        float v = acc[r] + bias;
        v = 0.5f * v * (1.f + erff(v * 0.70710678118f)); // exact GELU
        sH[(mt * 16 + r + hi * 8) * 96 + cl] = (_Float16)v;
      }
    }
    __syncthreads();
#pragma unroll
    for (int i = 0; i < 6; ++i) {
      const int gt = wv * 6 + i;       // 48 fc2 tiles
      const int mt = gt / 12, nt = gt % 12;
      for (int kk = 0; kk < 3; ++kk) {
        v16h a = ld_afrag(sH, 96, mt * 16, kk * 32, lane);
        v16h b = ld_bfrag(fc2T, 768, nt * 16, col0 + kk * 32, lane);
        facc[i] = wmma16(a, b, facc[i]);
      }
    }
    __syncthreads();
  }

#pragma unroll
  for (int i = 0; i < 6; ++i) {
    const int gt = wv * 6 + i;
    const int mt = gt / 12, nt = gt % 12;
    const int c = nt * 16 + (lane & 15);
    const float bias = fc2_b[c];
    const int hi = lane >> 4;
#pragma unroll
    for (int r = 0; r < 8; ++r) {
      const int row = mt * 16 + r + hi * 8;
      const int tok = tok0 + row;
      const int idx = (tok / 3136) * (DIMC * 3136) + c * 3136 + tok % 3136;
      out[idx] = x1[idx] + facc[i][r] + bias;
    }
  }
}

extern "C" void kernel_launch(void* const* d_in, const int* in_sizes, int n_in,
                              void* d_out, int out_size, void* d_ws, size_t ws_size,
                              hipStream_t stream) {
  const float* x      = (const float*)d_in[0];
  const float* n1g    = (const float*)d_in[1];
  const float* n1b    = (const float*)d_in[2];
  const float* qkv_w  = (const float*)d_in[3];
  const float* qkv_b  = (const float*)d_in[4];
  const float* proj_w = (const float*)d_in[5];
  const float* proj_b = (const float*)d_in[6];
  const float* relb   = (const float*)d_in[7];
  const float* n2g    = (const float*)d_in[8];
  const float* n2b    = (const float*)d_in[9];
  const float* fc1_w  = (const float*)d_in[10];
  const float* fc1_b  = (const float*)d_in[11];
  const float* fc2_w  = (const float*)d_in[12];
  const float* fc2_b  = (const float*)d_in[13];

  char* ws = (char*)d_ws;
  _Float16* qkvT = (_Float16*)ws;                 // [576][192]
  _Float16* projT = qkvT + 576 * 192;             // [192][192]
  _Float16* fc1T  = projT + 192 * 192;            // [768][192]
  _Float16* fc2T  = fc1T + 768 * 192;             // [192][768]
  const size_t wbytes = (size_t)(576 * 192 + 192 * 192 + 768 * 192 + 192 * 768) * 2;
  float* x1 = (float*)(ws + ((wbytes + 255) / 256) * 256); // [16][192][56][56] f32

  convert_weights_kernel<<<(768 * 192 + 255) / 256, 256, 0, stream>>>(
      qkv_w, proj_w, fc1_w, fc2_w, qkvT, projT, fc1T, fc2T);
  swin_attn_kernel<<<1024, 256, 0, stream>>>(x, n1g, n1b, qkvT, qkv_b, projT, proj_b, relb, x1);
  swin_mlp_kernel<<<784, 256, 0, stream>>>(x1, n2g, n2b, fc1T, fc1_b, fc2T, fc2_b, (float*)d_out);
}
